// ARCViT2_36610301231637
// MI455X (gfx1250) — compile-verified, matrix operand order checked by reading, not run on hardware
//
#include <hip/hip_runtime.h>
#include <hip/hip_bf16.h>
#include <math.h>

typedef __attribute__((ext_vector_type(16))) _Float16 v16h;
typedef __attribute__((ext_vector_type(8)))  float    v8f;

// ---------------- model constants ----------------
constexpr int E     = 768;
constexpr int NH    = 12;
constexpr int HD    = 64;          // E/NH
constexpr int LYR   = 6;
constexpr int G28   = 28;
constexpr int PP    = 2;
constexpr int NC    = 10;
constexpr int NT    = 1;
constexpr int S_IMG = G28 * G28;   // 784
constexpr int SEQ   = S_IMG + NT;  // 785
constexpr int SB    = 832;         // padded seq (multiple of 64)
constexpr int HID   = 2048;
constexpr int FC1N  = 2 * HID;     // 4096
constexpr int NRELD = 2 * G28 - 1; // 55
constexpr int NREL  = NRELD * NRELD;
constexpr float SCALE = 0.125f;    // HD^-0.5

// LDS A-tile row stride: 64 data halves + 8 pad halves (bank-conflict-free)
constexpr int AS_STRIDE = 72;      // halves (= 36 dwords = 9 uint4, 144 B)

// =====================================================================
// Batched WMMA GEMM:  C[m,n] = sum_k A[m,k] * Bm[n,k]  (+bias[n])
// A: f16 row-major (lda), Bm: f16 row-major along K (ldb), C: f32 (ldc)
// Block tile 32(M) x 64(N) x 64(K); 128 threads = 4 waves, each wave owns a
// 16-wide N strip and both M halves (2 acc tiles, 4 WMMAs per K chunk).
// LDS A-tile is double-buffered; one barrier per 64-K chunk.
// Requirements: M%32==0, N%64==0, K%64==0 (guaranteed by padding).
// grid: (N/64, M/32, Z); z1=z/zdiv, z2=z%zdiv select batch offsets.
// =====================================================================
__global__ __launch_bounds__(128)
void gemm_wmma_f16(const _Float16* __restrict__ Amat,
                   const _Float16* __restrict__ Bmat,
                   float* __restrict__ C,
                   const float* __restrict__ bias,
                   int N, int K, int lda, int ldb, int ldc, int zdiv,
                   long long sA1, long long sA2,
                   long long sB1, long long sB2,
                   long long sC1, long long sC2)
{
    __shared__ _Float16 As[2][32 * AS_STRIDE];

    const long long z1 = blockIdx.z / zdiv;
    const long long z2 = blockIdx.z % zdiv;
    Amat += z1 * sA1 + z2 * sA2;
    Bmat += z1 * sB1 + z2 * sB2;
    C    += z1 * sC1 + z2 * sC2;

    const int lane  = threadIdx.x & 31;
    const int wave  = threadIdx.x >> 5;
    const int mBase = blockIdx.y * 32;
    const int nCol  = blockIdx.x * 64 + wave * 16 + (lane & 15);
    const int hi    = lane >> 4;          // 0: K+0..7 half, 1: K+8..15 half
    const unsigned kb2 = hi * 4;          // dword offset of half

    // ---- cooperative A-tile staging: 32 rows x 64 halves = 256 uint4 ----
    // row = idx>>3 (8 uint4 of data per row), c4 = idx&7
    auto loadA = [&](int k0, uint4 rg[2]) {
        #pragma unroll
        for (int i = 0; i < 2; ++i) {
            int idx = threadIdx.x + i * 128;
            int row = idx >> 3, c4 = idx & 7;
            rg[i] = ((const uint4*)(Amat + (size_t)(mBase + row) * lda + k0))[c4];
        }
    };
    auto storeA = [&](int buf, const uint4 rg[2]) {
        #pragma unroll
        for (int i = 0; i < 2; ++i) {
            int idx = threadIdx.x + i * 128;
            int row = idx >> 3, c4 = idx & 7;
            ((uint4*)(As[buf] + (size_t)row * AS_STRIDE))[c4] = rg[i];
        }
    };

    v8f acc0 = {}, acc1 = {};
    uint4 rg[2];
    loadA(0, rg);
    storeA(0, rg);
    __syncthreads();

    int buf = 0;
    for (int k0 = 0; k0 < K; k0 += 64) {
        const bool haveNext = (k0 + 64 < K);
        if (haveNext) {
            loadA(k0 + 64, rg);                          // global -> regs
            __builtin_prefetch(Bmat + (size_t)nCol * ldb + k0 + 64, 0, 0);
        }
        const unsigned* Br = (const unsigned*)(Bmat + (size_t)nCol * ldb + k0);
        #pragma unroll
        for (int ks = 0; ks < 2; ++ks) {                 // two 32-K sub-steps
            union { v16h h; unsigned u[8]; } ub, ua0, ua1;
            const unsigned* Ar0 = (const unsigned*)As[buf] +
                                  (lane & 15) * (AS_STRIDE / 2) + ks * 16;
            const unsigned* Ar1 = Ar0 + 16 * (AS_STRIDE / 2);
            #pragma unroll
            for (int v = 0; v < 8; ++v) {
                // ISA 16-bit fragment K map: V0..3 -> K0..7(+hi*8),
                //                            V4..7 -> K16..23(+hi*8)
                unsigned dw = (v < 4) ? (kb2 + v) : (8 + kb2 + (v - 4));
                ub.u[v]  = Br[ks * 16 + dw];
                ua0.u[v] = Ar0[dw];
                ua1.u[v] = Ar1[dw];
            }
            acc0 = __builtin_amdgcn_wmma_f32_16x16x32_f16(
                false, ua0.h, false, ub.h, (short)0, acc0, false, false);
            acc1 = __builtin_amdgcn_wmma_f32_16x16x32_f16(
                false, ua1.h, false, ub.h, (short)0, acc1, false, false);
        }
        if (haveNext) storeA(buf ^ 1, rg);               // regs -> LDS (other buf)
        __syncthreads();
        buf ^= 1;
    }

    const float bv = bias ? bias[nCol] : 0.f;
    #pragma unroll
    for (int r = 0; r < 8; ++r) {
        int m = mBase + hi * 8 + r;                      // C layout: vgpr r
        C[(size_t)m * ldc + nCol]        = acc0[r] + bv;
        C[(size_t)(m + 16) * ldc + nCol] = acc1[r] + bv;
    }
}

// ============================ small kernels ============================

__global__ void f32_to_f16_kernel(const float* __restrict__ src,
                                  _Float16* __restrict__ dst, long long n)
{
    long long i = (long long)blockIdx.x * blockDim.x + threadIdx.x;
    long long stride = (long long)gridDim.x * blockDim.x;
    for (; i < n; i += stride) dst[i] = (_Float16)src[i];
}

// head weights: pad 40x768 -> 64x768 (zero rows)
__global__ void head_pad_kernel(const float* __restrict__ src,
                                _Float16* __restrict__ dst)
{
    int r = blockIdx.x;
    for (int e = threadIdx.x; e < E; e += blockDim.x)
        dst[(size_t)r * E + e] =
            (r < NC * PP * PP) ? (_Float16)src[(size_t)r * E + e] : (_Float16)0.f;
}

// pm[pp*NC+c][o] = sum_i color_embed[c][i] * patch_w[pp][i][o]
__global__ void precompute_pm_kernel(const float* __restrict__ color_embed,
                                     const float* __restrict__ patch_w,
                                     float* __restrict__ pm)
{
    int pc = blockIdx.x;                  // 0..39
    int pp = pc / NC, c = pc % NC;
    for (int o = threadIdx.x; o < E; o += blockDim.x) {
        float acc = 0.f;
        for (int i = 0; i < E; ++i)
            acc += color_embed[(size_t)c * E + i] *
                   patch_w[((size_t)pp * E + i) * E + o];
        pm[(size_t)pc * E + o] = acc;
    }
}

// build x[b, s, :]  (grid: (SB, B))
__global__ void build_x_kernel(const int* __restrict__ pix,
                               const int* __restrict__ task_ids,
                               const float* __restrict__ task_embed,
                               const float* __restrict__ pm,
                               const float* __restrict__ patch_b,
                               const float* __restrict__ pos_embed,
                               float* __restrict__ x)
{
    int s = blockIdx.x, b = blockIdx.y;
    float* xr = x + ((size_t)b * SB + s) * E;
    if (s >= SEQ) {
        for (int e = threadIdx.x; e < E; e += blockDim.x) xr[e] = 0.f;
        return;
    }
    if (s < NT) {
        int t = task_ids[b];
        for (int e = threadIdx.x; e < E; e += blockDim.x)
            xr[e] = task_embed[(size_t)t * E + e];
        return;
    }
    int si = s - NT, gy = si / G28, gx = si % G28;
    int p[4];
    #pragma unroll
    for (int py = 0; py < PP; ++py)
        #pragma unroll
        for (int px = 0; px < PP; ++px)
            p[py * PP + px] =
                pix[((size_t)b * (G28 * PP) + gy * PP + py) * (G28 * PP) +
                    gx * PP + px];
    for (int e = threadIdx.x; e < E; e += blockDim.x) {
        float v = patch_b[e] + pos_embed[(size_t)si * E + e];
        #pragma unroll
        for (int q = 0; q < 4; ++q)
            v += pm[((size_t)q * NC + p[q]) * E + e];
        xr[e] = v;
    }
}

// RMSNorm into f16 (one block per token row)
__global__ void rmsnorm_f16_kernel(const float* __restrict__ x,
                                   const float* __restrict__ w,
                                   _Float16* __restrict__ out)
{
    __shared__ float red[256];
    const float* xr = x + (size_t)blockIdx.x * E;
    float ss = 0.f;
    for (int e = threadIdx.x; e < E; e += 256) { float v = xr[e]; ss += v * v; }
    red[threadIdx.x] = ss; __syncthreads();
    for (int s = 128; s > 0; s >>= 1) {
        if (threadIdx.x < s) red[threadIdx.x] += red[threadIdx.x + s];
        __syncthreads();
    }
    float r = rsqrtf(red[0] / (float)E + 1e-6f);
    _Float16* orow = out + (size_t)blockIdx.x * E;
    for (int e = threadIdx.x; e < E; e += 256)
        orow[e] = (_Float16)(xr[e] * r * w[e]);
}

// qkv f32 -> q/k (rope, f16) and v transposed (f16). grid: (SB, NH, B), 64 thr
__global__ void rope_split_kernel(const float* __restrict__ qkv,
                                  _Float16* __restrict__ q,
                                  _Float16* __restrict__ k,
                                  _Float16* __restrict__ vt)
{
    int s = blockIdx.x, h = blockIdx.y, b = blockIdx.z, d = threadIdx.x;
    size_t z  = (size_t)b * NH + h;
    size_t qo = (z * SB + s) * HD + d;
    size_t vo = (z * HD + d) * SB + s;
    if (s >= SEQ) { q[qo] = (_Float16)0.f; k[qo] = (_Float16)0.f;
                    vt[vo] = (_Float16)0.f; return; }
    size_t ib = ((size_t)b * SB + s) * (3 * E) + (size_t)h * HD + d;
    float qv = qkv[ib], kv = qkv[ib + E], vv = qkv[ib + 2 * E];
    if (s >= NT) {
        int si = s - NT, py = si / G28, px = si % G28;
        int fi = (d & 31) >> 1;
        float freq = powf(10000.f, -(float)(2 * fi) / 32.f);
        float pos  = (d < 32) ? (float)py : (float)px;
        float ang  = pos * freq;
        float c = cosf(ang), sn = sinf(ang);
        size_t pb = ((size_t)b * SB + s) * (3 * E) + (size_t)h * HD + (d ^ 1);
        float qp = qkv[pb], kp = qkv[pb + E];
        float qr = (d & 1) ? qp : -qp;     // rotate_half
        float kr = (d & 1) ? kp : -kp;
        qv = qv * c + qr * sn;
        kv = kv * c + kr * sn;
    }
    q[qo] = (_Float16)qv; k[qo] = (_Float16)kv; vt[vo] = (_Float16)vv;
}

// softmax(scores*scale + relbias) -> f16 probs. grid: (SB, NH, B)
__global__ void softmax_bias_kernel(const float* __restrict__ sc,
                                    const float* __restrict__ btab,
                                    _Float16* __restrict__ attn)
{
    __shared__ float red[256];
    int qi = blockIdx.x, h = blockIdx.y, b = blockIdx.z;
    size_t base = (((size_t)b * NH + h) * SB + qi) * SB;
    _Float16* arow = attn + base;
    if (qi >= SEQ) {
        for (int kk = threadIdx.x; kk < SB; kk += 256) arow[kk] = (_Float16)0.f;
        return;
    }
    const float* srow = sc + base;
    int qy = 0, qx = 0; bool qimg = (qi >= NT);
    if (qimg) { int t = qi - NT; qy = t / G28; qx = t % G28; }

    float mx = -1e30f;
    for (int kk = threadIdx.x; kk < SEQ; kk += 256) {
        float v = srow[kk] * SCALE;
        if (qimg && kk >= NT) {
            int t = kk - NT;
            int dy = qy - t / G28 + (G28 - 1);
            int dx = qx - t % G28 + (G28 - 1);
            v += btab[(size_t)(dy * NRELD + dx) * NH + h];
        }
        mx = fmaxf(mx, v);
    }
    red[threadIdx.x] = mx; __syncthreads();
    for (int s = 128; s > 0; s >>= 1) {
        if (threadIdx.x < s)
            red[threadIdx.x] = fmaxf(red[threadIdx.x], red[threadIdx.x + s]);
        __syncthreads();
    }
    mx = red[0]; __syncthreads();

    float sum = 0.f;
    for (int kk = threadIdx.x; kk < SEQ; kk += 256) {
        float v = srow[kk] * SCALE;
        if (qimg && kk >= NT) {
            int t = kk - NT;
            int dy = qy - t / G28 + (G28 - 1);
            int dx = qx - t % G28 + (G28 - 1);
            v += btab[(size_t)(dy * NRELD + dx) * NH + h];
        }
        float e = expf(v - mx);
        arow[kk] = (_Float16)e;
        sum += e;
    }
    red[threadIdx.x] = sum; __syncthreads();
    for (int s = 128; s > 0; s >>= 1) {
        if (threadIdx.x < s) red[threadIdx.x] += red[threadIdx.x + s];
        __syncthreads();
    }
    float inv = 1.f / red[0];
    for (int kk = threadIdx.x; kk < SB; kk += 256)
        arow[kk] = (kk < SEQ) ? (_Float16)((float)arow[kk] * inv)
                              : (_Float16)0.f;
}

// f32 token buffer -> f16 (zero pad rows)
__global__ void tok_to_f16_kernel(const float* __restrict__ src,
                                  _Float16* __restrict__ dst, int cols)
{
    int t = blockIdx.x, s = t % SB;
    const float* sr = src + (size_t)t * cols;
    _Float16*    dr = dst + (size_t)t * cols;
    for (int e = threadIdx.x; e < cols; e += blockDim.x)
        dr[e] = (s < SEQ) ? (_Float16)sr[e] : (_Float16)0.f;
}

// x += tmp + bias  (valid tokens only; keeps pad rows zero)
__global__ void add_residual_kernel(float* __restrict__ x,
                                    const float* __restrict__ t,
                                    const float* __restrict__ bias)
{
    int tok = blockIdx.x, s = tok % SB;
    if (s >= SEQ) return;
    float* xr = x + (size_t)tok * E;
    const float* tr = t + (size_t)tok * E;
    for (int e = threadIdx.x; e < E; e += blockDim.x)
        xr[e] += tr[e] + bias[e];
}

// depthwise 3x3 conv on u branch + exact GELU * g  -> f16.  grid: (TOK, 8)
__global__ void dwconv_glu_kernel(const float* __restrict__ uv,
                                  const float* __restrict__ dww,
                                  const float* __restrict__ dwb,
                                  _Float16* __restrict__ hm)
{
    int t = blockIdx.x;
    int c = blockIdx.y * blockDim.x + threadIdx.x;   // channel
    int s = t % SB, b = t / SB;
    size_t o = (size_t)t * HID + c;
    if (s >= SEQ) { hm[o] = (_Float16)0.f; return; }
    float u;
    if (s < NT) {
        u = uv[(size_t)t * FC1N + c];                // task token: no conv
    } else {
        int si = s - NT, y = si / G28, x = si % G28;
        float acc = dwb[c];
        #pragma unroll
        for (int dy = -1; dy <= 1; ++dy) {
            int yy = y + dy;
            if (yy < 0 || yy >= G28) continue;
            #pragma unroll
            for (int dx = -1; dx <= 1; ++dx) {
                int xx = x + dx;
                if (xx < 0 || xx >= G28) continue;
                acc += uv[((size_t)b * SB + NT + yy * G28 + xx) * FC1N + c] *
                       dww[(size_t)((dy + 1) * 3 + (dx + 1)) * HID + c];
            }
        }
        u = acc;
    }
    float g  = uv[(size_t)t * FC1N + HID + c];
    float ge = 0.5f * u * (1.f + erff(u * 0.70710678f));
    hm[o] = (_Float16)(ge * g);
}

// headout (TOK x 64) -> logits (B,56,56,NC)
__global__ void write_out_kernel(const float* __restrict__ ho,
                                 const float* __restrict__ head_b,
                                 float* __restrict__ out, int total)
{
    int i = blockIdx.x * blockDim.x + threadIdx.x;
    if (i >= total) return;
    int c = i % NC, rest = i / NC;
    int col = rest % (G28 * PP); rest /= (G28 * PP);
    int row = rest % (G28 * PP); int b = rest / (G28 * PP);
    int gy = row >> 1, py = row & 1, gx = col >> 1, px = col & 1;
    size_t sidx = (size_t)b * SB + NT + gy * G28 + gx;
    int o = (py * PP + px) * NC + c;
    out[i] = ho[sidx * 64 + o] + head_b[o];
}

// =====================================================================
extern "C" void kernel_launch(void* const* d_in, const int* in_sizes, int n_in,
                              void* d_out, int out_size, void* d_ws, size_t ws_size,
                              hipStream_t stream)
{
    const int*   pix        = (const int*)  d_in[0];
    const int*   task_ids   = (const int*)  d_in[1];
    const float* color_emb  = (const float*)d_in[2];
    const float* task_emb   = (const float*)d_in[3];
    const float* patch_w    = (const float*)d_in[4];
    const float* patch_b    = (const float*)d_in[5];
    const float* pos_emb    = (const float*)d_in[6];
    const float* qkv_w      = (const float*)d_in[7];
    const float* qkv_b      = (const float*)d_in[8];
    const float* proj_w     = (const float*)d_in[9];
    const float* proj_b     = (const float*)d_in[10];
    const float* norm1_w    = (const float*)d_in[11];
    const float* norm2_w    = (const float*)d_in[12];
    const float* fc1_w      = (const float*)d_in[13];
    const float* fc1_b      = (const float*)d_in[14];
    const float* dw_w       = (const float*)d_in[15];
    const float* dw_b       = (const float*)d_in[16];
    const float* fc2_w      = (const float*)d_in[17];
    const float* fc2_b      = (const float*)d_in[18];
    const float* bias_table = (const float*)d_in[19];
    const float* fnorm_w    = (const float*)d_in[20];
    const float* head_w     = (const float*)d_in[21];
    const float* head_b     = (const float*)d_in[22];
    float* out = (float*)d_out;

    const int B   = in_sizes[0] / (G28 * PP * G28 * PP);   // 4
    const int TOK = B * SB;                                 // 3328

    // ---- workspace layout ----
    unsigned char* ws = (unsigned char*)d_ws;
    size_t off = 0;
    auto alloc = [&](size_t bytes) -> void* {
        off = (off + 255) & ~(size_t)255;
        void* p = ws + off; off += bytes; return p;
    };
    _Float16* WQKV = (_Float16*)alloc((size_t)LYR * 3 * E * E * 2);
    _Float16* WPRJ = (_Float16*)alloc((size_t)LYR * E * E * 2);
    _Float16* WFC1 = (_Float16*)alloc((size_t)LYR * FC1N * E * 2);
    _Float16* WFC2 = (_Float16*)alloc((size_t)LYR * E * HID * 2);
    _Float16* WHED = (_Float16*)alloc((size_t)64 * E * 2);
    float*    PM   = (float*)   alloc((size_t)4 * NC * E * 4);
    float*    X    = (float*)   alloc((size_t)TOK * E * 4);
    _Float16* HA   = (_Float16*)alloc((size_t)TOK * E * 2);
    float*    QKV  = (float*)   alloc((size_t)TOK * 3 * E * 4);
    _Float16* Qb   = (_Float16*)alloc((size_t)B * NH * SB * HD * 2);
    _Float16* Kb   = (_Float16*)alloc((size_t)B * NH * SB * HD * 2);
    _Float16* VT   = (_Float16*)alloc((size_t)B * NH * HD * SB * 2);
    float*    SC   = (float*)   alloc((size_t)B * NH * SB * SB * 4);
    _Float16* AT   = (_Float16*)alloc((size_t)B * NH * SB * SB * 2);
    float*    CTX  = (float*)   alloc((size_t)TOK * E * 4);
    _Float16* CTXH = (_Float16*)alloc((size_t)TOK * E * 2);
    float*    TMP  = (float*)   alloc((size_t)TOK * FC1N * 4);
    _Float16* HM   = (_Float16*)alloc((size_t)TOK * HID * 2);
    float*    HO   = (float*)   alloc((size_t)TOK * 64 * 4);

    auto cvt = [&](const float* s, _Float16* d, long long n) {
        int blocks = (int)((n + 255 * 8) / (256 * 8));
        if (blocks > 4096) blocks = 4096;
        f32_to_f16_kernel<<<blocks, 256, 0, stream>>>(s, d, n);
    };

    // ---- weight down-conversion (f32 -> f16, once per call) ----
    cvt(qkv_w, WQKV, (long long)LYR * 3 * E * E);
    cvt(proj_w, WPRJ, (long long)LYR * E * E);
    cvt(fc1_w, WFC1, (long long)LYR * FC1N * E);
    cvt(fc2_w, WFC2, (long long)LYR * E * HID);
    head_pad_kernel<<<64, 256, 0, stream>>>(head_w, WHED);

    // ---- embedding ----
    precompute_pm_kernel<<<4 * NC, 256, 0, stream>>>(color_emb, patch_w, PM);
    build_x_kernel<<<dim3(SB, B), 256, 0, stream>>>(
        pix, task_ids, task_emb, PM, patch_b, pos_emb, X);

    const int mt = TOK / 32;   // token-row tiles of 32 (104)
    for (int l = 0; l < LYR; ++l) {
        // --- attention ---
        rmsnorm_f16_kernel<<<TOK, 256, 0, stream>>>(X, norm1_w + (size_t)l * E, HA);
        gemm_wmma_f16<<<dim3(3 * E / 64, mt, 1), 128, 0, stream>>>(
            HA, WQKV + (size_t)l * 3 * E * E, QKV, qkv_b + (size_t)l * 3 * E,
            3 * E, E, E, E, 3 * E, 1, 0, 0, 0, 0, 0, 0);
        rope_split_kernel<<<dim3(SB, NH, B), HD, 0, stream>>>(QKV, Qb, Kb, VT);
        // scores: per (b,h)  [SB x SB] = q [SB x 64] * k^T
        gemm_wmma_f16<<<dim3(SB / 64, SB / 32, B * NH), 128, 0, stream>>>(
            Qb, Kb, SC, nullptr, SB, HD, HD, HD, SB, 1,
            (long long)SB * HD, 0, (long long)SB * HD, 0,
            (long long)SB * SB, 0);
        softmax_bias_kernel<<<dim3(SB, NH, B), 256, 0, stream>>>(
            SC, bias_table + (size_t)l * NREL * NH, AT);
        // ctx: per (b,h)  [SB x 64] = attn [SB x SB] * v ; C -> (b, s, h*HD+d)
        gemm_wmma_f16<<<dim3(1, SB / 32, B * NH), 128, 0, stream>>>(
            AT, VT, CTX, nullptr, HD, SB, SB, SB, E, NH,
            (long long)NH * SB * SB, (long long)SB * SB,
            (long long)NH * HD * SB, (long long)HD * SB,
            (long long)SB * E, (long long)HD);
        tok_to_f16_kernel<<<TOK, 256, 0, stream>>>(CTX, CTXH, E);
        gemm_wmma_f16<<<dim3(E / 64, mt, 1), 128, 0, stream>>>(
            CTXH, WPRJ + (size_t)l * E * E, TMP, nullptr,
            E, E, E, E, E, 1, 0, 0, 0, 0, 0, 0);
        add_residual_kernel<<<TOK, 256, 0, stream>>>(X, TMP, proj_b + (size_t)l * E);

        // --- MLP (SwiGLU + depthwise conv) ---
        rmsnorm_f16_kernel<<<TOK, 256, 0, stream>>>(X, norm2_w + (size_t)l * E, HA);
        gemm_wmma_f16<<<dim3(FC1N / 64, mt, 1), 128, 0, stream>>>(
            HA, WFC1 + (size_t)l * FC1N * E, TMP, fc1_b + (size_t)l * FC1N,
            FC1N, E, E, E, FC1N, 1, 0, 0, 0, 0, 0, 0);
        dwconv_glu_kernel<<<dim3(TOK, HID / 256), 256, 0, stream>>>(
            TMP, dw_w + (size_t)l * 9 * HID, dw_b + (size_t)l * HID, HM);
        gemm_wmma_f16<<<dim3(E / 64, mt, 1), 128, 0, stream>>>(
            HM, WFC2 + (size_t)l * E * HID, TMP, nullptr,
            E, HID, HID, HID, E, 1, 0, 0, 0, 0, 0, 0);
        add_residual_kernel<<<TOK, 256, 0, stream>>>(X, TMP, fc2_b + (size_t)l * E);
    }

    // ---- head ----
    rmsnorm_f16_kernel<<<TOK, 256, 0, stream>>>(X, fnorm_w, HA);
    gemm_wmma_f16<<<dim3(1, mt, 1), 128, 0, stream>>>(
        HA, WHED, HO, nullptr, 64, E, E, E, 64, 1, 0, 0, 0, 0, 0, 0);
    const int total = B * G28 * PP * G28 * PP * NC;
    write_out_kernel<<<(total + 255) / 256, 256, 0, stream>>>(HO, head_b, out, total);
}
